// ModelRNNv3_54193897341085
// MI455X (gfx1250) — matrix operand order repaired and checked by reading, not compiled, not run on hardware
//
#include <hip/hip_runtime.h>
#include <math.h>

// Problem constants (match reference)
#define B_SZ  256
#define T_SZ  1024
#define H_SZ  512
#define NIN   64
#define NOUT  64

#define MT    16            // batch-tile rows (WMMA M)
#define RBLK  512           // recurrence block: 16 waves, each owns 32 output cols
#define HP    (H_SZ + 4)    // f32 LDS row stride: 516 dw == 4 mod 64 banks
#define HPB   (H_SZ + 8)    // bf16 LDS row stride: 1040 B (16B-aligned, 260 dw == 4 mod 64)

typedef float  v2f  __attribute__((ext_vector_type(2)));
typedef float  v8f  __attribute__((ext_vector_type(8)));
typedef __bf16 v8bf __attribute__((ext_vector_type(8)));
typedef __bf16 v16bf __attribute__((ext_vector_type(16)));

// D = A(16x4 f32) * B(4x16 f32) + C  -- exact fp32 matrix pipe
static __device__ __forceinline__ v8f wmma_f32(v2f a, v2f b, v8f c) {
  return __builtin_amdgcn_wmma_f32_16x16x4_f32(false, a, false, b, (short)0, c, false, false);
}
// D = A(16x32 bf16) * B(32x16 bf16) + C(f32)  -- 8x K per instruction
static __device__ __forceinline__ v8f wmma_bf16(v16bf a, v16bf b, v8f c) {
  return __builtin_amdgcn_wmma_f32_16x16x32_bf16(false, a, false, b, (short)0, c, false, false);
}
// assemble a 16-element bf16 operand from two 16-byte chunks
static __device__ __forceinline__ v16bf ld2x8(const __bf16* p0, const __bf16* p1) {
  v8bf a = *(const v8bf*)p0;
  v8bf b = *(const v8bf*)p1;
  return __builtin_shufflevector(a, b, 0,1,2,3,4,5,6,7,8,9,10,11,12,13,14,15);
}
// fast tanh: 1 - 2*rcp(exp(2x)+1); exact +/-1 saturation, v_exp + v_rcp
static __device__ __forceinline__ float fast_tanh(float x) {
  float e = __expf(2.0f * x);
  return 1.0f - 2.0f * __builtin_amdgcn_rcpf(e + 1.0f);
}

// ---------------------------------------------------------------------------
// Prep: split W (f32) into hi/lo bf16 planes (bf16x3 scheme), stored in d_ws.
// ---------------------------------------------------------------------------
__global__ __launch_bounds__(256)
void prep_w_bf16(const float* __restrict__ W,
                 __bf16* __restrict__ whi, __bf16* __restrict__ wlo) {
  int i = blockIdx.x * 256 + threadIdx.x;
  float w = W[i];
  __bf16 h = (__bf16)w;
  whi[i] = h;
  wlo[i] = (__bf16)(w - (float)h);
}

// ---------------------------------------------------------------------------
// Input projection: inp[r, :] = X[r, :] @ Win^T  (M = B*T, N = 512, K = 64)
// Written INTO the hidden region of d_out (recurrence consumes inp[b,t,:] as
// its C-initializer, then overwrites the same addresses with h_next).
// Exact fp32 WMMA; massively parallel, off the serial path.
// ---------------------------------------------------------------------------
__global__ __launch_bounds__(256)
void rnn_inproj_kernel(const float* __restrict__ X,    // [B*T, NIN]
                       const float* __restrict__ Win,  // [H, NIN]
                       float* __restrict__ inp)        // [B*T, H] (= hidden region)
{
  const int tid  = threadIdx.x;
  const int wave = tid >> 5;         // 8 waves, each 64 output cols
  const int lane = tid & 31;
  const int mi   = lane & 15;
  const int kh   = (lane >> 4) << 1;
  const int mb   = (lane >> 4) << 3;
  const size_t r0 = (size_t)blockIdx.x * 16;   // one 16-row slab per block
  const int c0 = wave * 64;

  const float* ar = X + (r0 + mi) * NIN;
  const float* wr = Win + (size_t)(c0 + mi) * NIN;

  v8f acc[4] = {{}, {}, {}, {}};
  #pragma unroll
  for (int k0 = 0; k0 < NIN; k0 += 4) {
    v2f a = *(const v2f*)(ar + k0 + kh);
    #pragma unroll
    for (int j = 0; j < 4; ++j) {
      v2f b = *(const v2f*)(wr + (size_t)j * 16 * NIN + k0 + kh);
      acc[j] = wmma_f32(a, b, acc[j]);
    }
  }
  #pragma unroll
  for (int j = 0; j < 4; ++j)
    #pragma unroll
    for (int r = 0; r < 8; ++r)
      inp[(r0 + mb + r) * H_SZ + c0 + j * 16 + mi] = acc[j][r];
}

// ---------------------------------------------------------------------------
// Recurrence (bf16x3): one persistent block per 16-row batch tile, all T steps.
// Step t: acc <- inp tile (loaded from hidden[.,t,:], pre-warmed into L2 by a
// prefetch issued at step t-1), then acc += tanh(h) @ W^T as Ah*Bh+Ah*Bl+Al*Bh
// in bf16 WMMA with f32 accumulate; h_next overwrites the inp tile in place.
// ---------------------------------------------------------------------------
__global__ __launch_bounds__(RBLK)
void rnn_recurrence_bf16x3(const float* __restrict__ h0,     // [B,H]
                           const __bf16* __restrict__ Whi,   // [H,H]
                           const __bf16* __restrict__ Wlo,   // [H,H]
                           float* __restrict__ hidden)       // [B,T,H] (inp on entry)
{
  __shared__ __bf16 thi[MT][HPB];   // tanh(h) hi plane
  __shared__ __bf16 tlo[MT][HPB];   // tanh(h) lo plane

  const int tid  = threadIdx.x;
  const int b0   = blockIdx.x * MT;
  const int wave = tid >> 5;
  const int lane = tid & 31;
  const int mi   = lane & 15;          // A row / B,C column within tile
  const int lh   = lane >> 4;          // K half selector
  const int n0   = wave * 32;          // this wave's first output column
  const int mb   = lh << 3;            // C row base: 0 or 8
  const int nc   = n0 + mi;

  // tbuf = split(tanh(h0))
  for (int i = tid; i < MT * H_SZ; i += RBLK) {
    int m = i >> 9;
    int c = i & (H_SZ - 1);
    float tv = fast_tanh(h0[(size_t)(b0 + m) * H_SZ + c]);
    __bf16 hv = (__bf16)tv;
    thi[m][c] = hv;
    tlo[m][c] = (__bf16)(tv - (float)hv);
  }
  __syncthreads();

  const __bf16* wh0 = Whi + (size_t)(n0 + mi) * H_SZ;
  const __bf16* wh1 = Whi + (size_t)(n0 + 16 + mi) * H_SZ;
  const __bf16* wl0 = Wlo + (size_t)(n0 + mi) * H_SZ;
  const __bf16* wl1 = Wlo + (size_t)(n0 + 16 + mi) * H_SZ;
  const __bf16* ahr = &thi[mi][0];
  const __bf16* alr = &tlo[mi][0];

  // one 128B line per lane covers this wave's whole inp slice of a row
  const float* pfb = hidden + (size_t)(b0 + mi) * T_SZ * H_SZ + n0 + lh * 16;

  for (int t = 0; t < T_SZ; ++t) {
    // --- C-init: load this step's input-projection tile (L2-warm) ---
    v8f acc0, acc1;
    #pragma unroll
    for (int r = 0; r < 8; ++r) {
      size_t g = ((size_t)(b0 + mb + r) * T_SZ + t) * H_SZ;
      acc0[r] = hidden[g + nc];
      acc1[r] = hidden[g + nc + 16];
    }

    // --- prefetch next step's inp tile into cache (no LOADcnt impact) ---
    if (t + 1 < T_SZ)
      __builtin_prefetch(pfb + (size_t)(t + 1) * H_SZ, 0, 1);

    // --- recurrent GEMM: tanh(h) @ W^T via bf16x3 (K = 512, 32/chunk) ---
    #pragma unroll 4
    for (int kc = 0; kc < H_SZ; kc += 32) {
      const int aoff = kc + 16 * lh;      // A: K sub-block per lane half
      const int boff = kc + 16 * lh;      // B: K sub-block per lane half
      v16bf ah  = ld2x8(ahr + aoff, ahr + aoff + 8);
      v16bf al  = ld2x8(alr + aoff, alr + aoff + 8);
      v16bf bh0 = ld2x8(wh0 + boff, wh0 + boff + 8);
      v16bf bl0 = ld2x8(wl0 + boff, wl0 + boff + 8);
      v16bf bh1 = ld2x8(wh1 + boff, wh1 + boff + 8);
      v16bf bl1 = ld2x8(wl1 + boff, wl1 + boff + 8);
      acc0 = wmma_bf16(ah, bh0, acc0);
      acc0 = wmma_bf16(ah, bl0, acc0);
      acc0 = wmma_bf16(al, bh0, acc0);
      acc1 = wmma_bf16(ah, bh1, acc1);
      acc1 = wmma_bf16(ah, bl1, acc1);
      acc1 = wmma_bf16(al, bh1, acc1);
    }

    // --- overwrite inp tile with raw h_next (f32) ---
    #pragma unroll
    for (int r = 0; r < 8; ++r) {
      size_t g = ((size_t)(b0 + mb + r) * T_SZ + t) * H_SZ;
      hidden[g + nc]      = acc0[r];
      hidden[g + nc + 16] = acc1[r];
    }

    __syncthreads();   // all waves done reading thi/tlo for this step

    // --- write split(tanh(h_next)) tile back into LDS ---
    #pragma unroll
    for (int r = 0; r < 8; ++r) {
      int m = mb + r;
      float t0 = fast_tanh(acc0[r]);
      float t1 = fast_tanh(acc1[r]);
      __bf16 h0b = (__bf16)t0;
      __bf16 h1b = (__bf16)t1;
      thi[m][nc]      = h0b;
      tlo[m][nc]      = (__bf16)(t0 - (float)h0b);
      thi[m][nc + 16] = h1b;
      tlo[m][nc + 16] = (__bf16)(t1 - (float)h1b);
    }
    __syncthreads();   // ready for step t+1
  }
}

// ---------------------------------------------------------------------------
// Fallback all-fp32 recurrence (used only if d_ws cannot hold W_hi/W_lo).
// ---------------------------------------------------------------------------
__global__ __launch_bounds__(RBLK)
void rnn_recurrence_f32(const float* __restrict__ X,
                        const float* __restrict__ h0,
                        const float* __restrict__ W,
                        const float* __restrict__ Win,
                        float* __restrict__ hidden)
{
  __shared__ float tbuf[MT][HP];

  const int tid  = threadIdx.x;
  const int b0   = blockIdx.x * MT;
  const int wave = tid >> 5;
  const int lane = tid & 31;
  const int mi   = lane & 15;
  const int kh   = (lane >> 4) << 1;
  const int n0   = wave * 32;
  const int mb   = (lane >> 4) << 3;

  for (int i = tid; i < MT * H_SZ; i += RBLK) {
    int m = i >> 9;
    int c = i & (H_SZ - 1);
    tbuf[m][c] = fast_tanh(h0[(size_t)(b0 + m) * H_SZ + c]);
  }
  __syncthreads();

  const float* xbase = X + (size_t)(b0 + mi) * T_SZ * NIN;
  const float* wi0 = Win + (size_t)(n0 + mi) * NIN;
  const float* wi1 = Win + (size_t)(n0 + 16 + mi) * NIN;
  const float* w0  = W + (size_t)(n0 + mi) * H_SZ;
  const float* w1  = W + (size_t)(n0 + 16 + mi) * H_SZ;

  for (int t = 0; t < T_SZ; ++t) {
    v8f acc0 = {};
    v8f acc1 = {};
    {
      const float* xr = xbase + (size_t)t * NIN;
      #pragma unroll
      for (int k0 = 0; k0 < NIN; k0 += 4) {
        v2f a  = *(const v2f*)(xr  + k0 + kh);
        v2f b0 = *(const v2f*)(wi0 + k0 + kh);
        v2f b1 = *(const v2f*)(wi1 + k0 + kh);
        acc0 = wmma_f32(a, b0, acc0);
        acc1 = wmma_f32(a, b1, acc1);
      }
    }
    {
      const float* ar = &tbuf[mi][0];
      #pragma unroll 4
      for (int k0 = 0; k0 < H_SZ; k0 += 4) {
        v2f a  = *(const v2f*)(ar + k0 + kh);
        v2f b0 = *(const v2f*)(w0 + k0 + kh);
        v2f b1 = *(const v2f*)(w1 + k0 + kh);
        acc0 = wmma_f32(a, b0, acc0);
        acc1 = wmma_f32(a, b1, acc1);
      }
    }
    const int nc = n0 + mi;
    #pragma unroll
    for (int r = 0; r < 8; ++r) {
      int m = mb + r;
      size_t g = ((size_t)(b0 + m) * T_SZ + t) * H_SZ;
      hidden[g + nc]      = acc0[r];
      hidden[g + nc + 16] = acc1[r];
    }
    __syncthreads();
    #pragma unroll
    for (int r = 0; r < 8; ++r) {
      int m = mb + r;
      tbuf[m][nc]      = fast_tanh(acc0[r]);
      tbuf[m][nc + 16] = fast_tanh(acc1[r]);
    }
    __syncthreads();
  }
}

// ---------------------------------------------------------------------------
// Output projection: out[r,:] = hidden[r,:] @ Wout^T  (M = B*T, N=64, K=512)
// Memory-bound (537 MB of hidden @ 23.3 TB/s); exact fp32 WMMA.
// ---------------------------------------------------------------------------
#define OBLK 256
__global__ __launch_bounds__(OBLK)
void rnn_outproj_kernel(const float* __restrict__ hidden, // [B*T, H]
                        const float* __restrict__ Wout,   // [NOUT, H]
                        float* __restrict__ out)          // [B*T, NOUT]
{
  const int tid  = threadIdx.x;
  const int wave = tid >> 5;
  const int lane = tid & 31;
  const int mi   = lane & 15;
  const int kh   = (lane >> 4) << 1;
  const int mb   = (lane >> 4) << 3;
  const size_t r0 = (size_t)blockIdx.x * 128 + (size_t)wave * 16;

  const float* ar = hidden + (size_t)(r0 + mi) * H_SZ;
  const float* wr = Wout + (size_t)mi * H_SZ;

  v8f acc[4] = {{}, {}, {}, {}};

  #pragma unroll 2
  for (int k0 = 0; k0 < H_SZ; k0 += 4) {
    v2f a = *(const v2f*)(ar + k0 + kh);
    #pragma unroll
    for (int j = 0; j < 4; ++j) {
      v2f b = *(const v2f*)(wr + (size_t)j * 16 * H_SZ + k0 + kh);
      acc[j] = wmma_f32(a, b, acc[j]);
    }
  }

  #pragma unroll
  for (int j = 0; j < 4; ++j) {
    #pragma unroll
    for (int r = 0; r < 8; ++r) {
      int m = mb + r;
      out[(r0 + m) * NOUT + (size_t)j * 16 + mi] = acc[j][r];
    }
  }
}

// ---------------------------------------------------------------------------
extern "C" void kernel_launch(void* const* d_in, const int* in_sizes, int n_in,
                              void* d_out, int out_size, void* d_ws, size_t ws_size,
                              hipStream_t stream) {
  (void)in_sizes; (void)n_in; (void)out_size;

  const float* X    = (const float*)d_in[0];
  const float* h0   = (const float*)d_in[1];
  const float* W    = (const float*)d_in[2];
  const float* Win  = (const float*)d_in[3];
  const float* Wout = (const float*)d_in[4];

  float* out    = (float*)d_out;                        // [B,T,NOUT]
  float* hidden = out + (size_t)B_SZ * T_SZ * NOUT;     // [B,T,H]

  const size_t wbytes = (size_t)H_SZ * H_SZ * sizeof(__bf16);  // 512 KB per plane

  if (d_ws != nullptr && ws_size >= 2 * wbytes) {
    __bf16* whi = (__bf16*)d_ws;
    __bf16* wlo = whi + (size_t)H_SZ * H_SZ;
    // 1) split W into bf16 hi/lo planes (L2-resident, 1 MB)
    prep_w_bf16<<<(H_SZ * H_SZ) / 256, 256, 0, stream>>>(W, whi, wlo);
    // 2) parallel input projection, written into the hidden region
    rnn_inproj_kernel<<<(B_SZ * T_SZ) / 16, 256, 0, stream>>>(X, Win, hidden);
    // 3) serial-in-T recurrence (bf16x3), consumes & overwrites inp in place
    rnn_recurrence_bf16x3<<<B_SZ / MT, RBLK, 0, stream>>>(h0, whi, wlo, hidden);
  } else {
    rnn_recurrence_f32<<<B_SZ / MT, RBLK, 0, stream>>>(X, h0, W, Win, hidden);
  }

  // 4) big parallel output GEMM
  rnn_outproj_kernel<<<(B_SZ * T_SZ) / 128, OBLK, 0, stream>>>(hidden, Wout, out);
}
